// PosAttDiff_64510408785999
// MI455X (gfx1250) — matrix operand order repaired and checked by reading, not compiled, not run on hardware
//
#include <hip/hip_runtime.h>

// ---------------------------------------------------------------------------
// MI455X (gfx1250) PosAttDiff:
//   out = (alpha*f1 + beta*f2) @ |softmax(a1^T b2) - softmax(a2^T b1)|^T
// All GEMMs via v_wmma_f32_16x16x32_bf16 (wave32, f32 accumulate).
// LDS tiles laid out so every WMMA fragment is 2x ds_load_b128 (no scalar DS).
// ---------------------------------------------------------------------------

typedef __bf16 bf16;
typedef __attribute__((ext_vector_type(16))) __bf16 v16bf;
typedef __attribute__((ext_vector_type(4)))  __bf16 v4bf;
typedef __attribute__((ext_vector_type(8)))  float  v8f;

#define WMMA_BF16(a, b, c) \
  __builtin_amdgcn_wmma_f32_16x16x32_bf16(false, (a), false, (b), (short)0, (c), false, false)

union Q4 { unsigned long long u; bf16 h[4]; };
union Q8 { uint4 u; bf16 h[8]; };
union F16x2 { uint4 u[2]; v16bf v; };

// ---- WMMA fragment loads (layouts per CDNA5 ISA §7.12.2) ----
// A 16x32 bf16 from LDS [m][k]: lane half=lane>>4, M=lane&15;
// elem e needs K = (e>>3)*16 + half*8 + (e&7)  ->  two contiguous 16B chunks.
static __device__ inline v16bf frag_a_mk(const bf16* s, int stride, int m_base,
                                         int k_base, int lane) {
  const int half = lane >> 4, mr = lane & 15;
  const bf16* p = s + (size_t)(m_base + mr) * stride + k_base + half * 8;
  F16x2 u;
  u.u[0] = *(const uint4*)(p);
  u.u[1] = *(const uint4*)(p + 16);
  return u.v;
}
// B 32x16 bf16 from LDS stored [n][k]: lane N=lane&15; elem e: K = (e>>3)*16 + half*8 + (e&7)
// -> two contiguous 16B chunks: two ds_load_b128.
static __device__ inline v16bf frag_b_nk(const bf16* s, int stride, int k_base,
                                         int n_base, int lane) {
  const int half = lane >> 4, nc = lane & 15;
  const bf16* p = s + (size_t)(n_base + nc) * stride + k_base + half * 8;
  F16x2 u;
  u.u[0] = *(const uint4*)(p);
  u.u[1] = *(const uint4*)(p + 16);
  return u.v;
}
// A fragment when logical A[m][k] is stored column-major s[k*stride+m]
// (used once per attention block to build the resident query fragments).
static __device__ inline v16bf frag_a_colmajor(const bf16* s, int stride, int m_base,
                                               int k_base, int lane) {
  const int half = lane >> 4, mr = lane & 15;
  v16bf f;
#pragma unroll
  for (int e = 0; e < 16; ++e) {
    const int k = k_base + (e >> 3) * 16 + half * 8 + (e & 7);
    f[e] = s[(size_t)k * stride + m_base + mr];
  }
  return f;
}

// ---------------------------------------------------------------------------
// pack: x1,x2 -> bf16; xc = alpha*x1 + beta*x2 -> bf16 (4 elems / thread)
// ---------------------------------------------------------------------------
__global__ __launch_bounds__(256) void pack_kernel(
    const float* __restrict__ x1, const float* __restrict__ x2,
    const float* __restrict__ alpha, const float* __restrict__ beta,
    bf16* __restrict__ x1b, bf16* __restrict__ x2b, bf16* __restrict__ xc, int total4) {
  const int i = blockIdx.x * blockDim.x + threadIdx.x;
  if (i >= total4) return;
  const float a = alpha[0], b = beta[0];
  const float4 v1 = ((const float4*)x1)[i];
  const float4 v2 = ((const float4*)x2)[i];
  v4bf o1 = {(bf16)v1.x, (bf16)v1.y, (bf16)v1.z, (bf16)v1.w};
  v4bf o2 = {(bf16)v2.x, (bf16)v2.y, (bf16)v2.z, (bf16)v2.w};
  v4bf oc = {(bf16)(a * v1.x + b * v2.x), (bf16)(a * v1.y + b * v2.y),
             (bf16)(a * v1.z + b * v2.z), (bf16)(a * v1.w + b * v2.w)};
  ((v4bf*)x1b)[i] = o1;
  ((v4bf*)x2b)[i] = o2;
  ((v4bf*)xc)[i]  = oc;
}

// ---------------------------------------------------------------------------
// weight/bias prep: Wab = [Wb;Wc] bf16, Wd bf16, bias_ab = [bb;bc], bias_d = (a+b)*bd
// ---------------------------------------------------------------------------
__global__ __launch_bounds__(256) void prep_weights(
    const float* __restrict__ Wb, const float* __restrict__ Wc, const float* __restrict__ Wdd,
    const float* __restrict__ bb, const float* __restrict__ bc, const float* __restrict__ bd,
    const float* __restrict__ alpha, const float* __restrict__ beta,
    bf16* __restrict__ Wab, bf16* __restrict__ Wd,
    float* __restrict__ bias_ab, float* __restrict__ bias_d) {
  const int i = blockIdx.x * blockDim.x + threadIdx.x;
  const int WABN = 128 * 512, WDN = 512 * 512;
  if (i < WABN) Wab[i] = (bf16)((i < 64 * 512) ? Wb[i] : Wc[i - 64 * 512]);
  if (i < WDN) Wd[i] = (bf16)Wdd[i];
  if (i < 128) bias_ab[i] = (i < 64) ? bb[i] : bc[i - 64];
  if (i < 512) bias_d[i] = (alpha[0] + beta[0]) * bd[i];
}

// ---------------------------------------------------------------------------
// 128x128-tile WMMA GEMM, BK=32. 256 threads = 8 waves (4M x 2N),
// each wave: 32x64 output = 2x4 16x16 WMMA tiles.
// LDS: As [m][k] (stride 40), BsT [n][k] (stride 40).
//   BT=false: B row-major [K x N] -> per-thread 4x4 register micro-transpose
//   BT=true : B given as att[n][k] -> straight vector copy
// ---------------------------------------------------------------------------
template <bool BT, typename OutT>
__global__ __launch_bounds__(256) void gemm128(
    const bf16* __restrict__ A, int lda, const bf16* __restrict__ B, int ldb,
    OutT* __restrict__ D, int ldd, int K, const float* __restrict__ bias) {
  constexpr int BM = 128, BN = 128, BK = 32;
  constexpr int AST = BK + 8;   // 40
  constexpr int BST = BK + 8;   // 40
  __shared__ bf16 As[BM * AST];
  __shared__ bf16 BsT[BN * BST];

  const int t = threadIdx.x, lane = t & 31, wid = t >> 5;
  const int wm = (wid >> 1) * 32, wn = (wid & 1) * 64;
  const int mblk = blockIdx.y * BM, nblk = blockIdx.x * BN;

  v8f acc[2][4];
#pragma unroll
  for (int i = 0; i < 2; ++i)
#pragma unroll
    for (int j = 0; j < 4; ++j) acc[i][j] = (v8f)(0.0f);

  // A: thread copies 32B of one row: m = t>>1, k-part = (t&1)*16
  const int am = t >> 1, akp = (t & 1) * 16;
  const bf16* ag = A + (size_t)(mblk + am) * lda + akp;
  // B (BT=false): 4k x 4n micro-tile: kg = t>>5 (0..7), ng = t&31 (0..31)
  const int kg = t >> 5, ng = t & 31;
  // B (BT=true): straight copy: n = t>>1, k-part = (t&1)*16
  const int bn = t >> 1, bkp = (t & 1) * 16;
  const bf16* bg = BT ? (B + (size_t)(nblk + bn) * ldb + bkp)
                      : (B + (size_t)(kg * 4) * ldb + nblk + ng * 4);

  const int ksteps = K / BK;
  for (int ks = 0; ks < ksteps; ++ks) {
    {  // A tile -> LDS (2x b128)
      const uint4* src = (const uint4*)(ag + (size_t)ks * BK);
      uint4 d0 = src[0], d1 = src[1];
      uint4* dst = (uint4*)(&As[am * AST + akp]);
      dst[0] = d0; dst[1] = d1;
    }
    if constexpr (BT) {  // att[n][k] -> BsT[n][k] (2x b128)
      const uint4* src = (const uint4*)(bg + (size_t)ks * BK);
      uint4 d0 = src[0], d1 = src[1];
      uint4* dst = (uint4*)(&BsT[bn * BST + bkp]);
      dst[0] = d0; dst[1] = d1;
    } else {  // B[k][n] -> BsT[n][k]: 4x b64 coalesced loads, 4x4 reg transpose, 4x ds_store_b64
      Q4 r[4];
#pragma unroll
      for (int i = 0; i < 4; ++i)
        r[i].u = *(const unsigned long long*)(bg + ((size_t)ks * BK + i) * ldb);
#pragma unroll
      for (int j = 0; j < 4; ++j) {
        Q4 c;
#pragma unroll
        for (int i = 0; i < 4; ++i) c.h[i] = r[i].h[j];
        *(unsigned long long*)(&BsT[(ng * 4 + j) * BST + kg * 4]) = c.u;
      }
    }
    if (ks + 1 < ksteps) {  // global_prefetch_b8 hints for next tiles
      __builtin_prefetch(ag + (size_t)(ks + 1) * BK, 0, 1);
      __builtin_prefetch(BT ? (bg + (size_t)(ks + 1) * BK)
                            : (bg + (size_t)(ks + 1) * BK * ldb), 0, 1);
    }
    __syncthreads();

    v16bf fa[2], fb[4];
#pragma unroll
    for (int i = 0; i < 2; ++i) fa[i] = frag_a_mk(As, AST, wm + i * 16, 0, lane);
#pragma unroll
    for (int j = 0; j < 4; ++j) fb[j] = frag_b_nk(BsT, BST, 0, wn + j * 16, lane);
#pragma unroll
    for (int i = 0; i < 2; ++i)
#pragma unroll
      for (int j = 0; j < 4; ++j) acc[i][j] = WMMA_BF16(fa[i], fb[j], acc[i][j]);
    __syncthreads();
  }

  // epilogue: C/D layout — VGPR r holds rows M=r (lanes 0-15) and M=r+8 (lanes 16-31)
  const int half = lane >> 4, nc = lane & 15;
#pragma unroll
  for (int i = 0; i < 2; ++i)
#pragma unroll
    for (int j = 0; j < 4; ++j)
#pragma unroll
      for (int r = 0; r < 8; ++r) {
        const int row = mblk + wm + i * 16 + r + 8 * half;
        const int col = nblk + wn + j * 16 + nc;
        float v = acc[i][j][r];
        if (bias) v += bias[row];
        D[(size_t)row * ldd + col] = (OutT)v;
      }
}

// ---------------------------------------------------------------------------
// Fused attention: per 128-query block, flash-style two passes over m.
//   S12 = a1^T b2, S21 = a2^T b1 (K=64, 2 WMMA steps).
// Pass 1: LANE-LOCAL online max/sum (no cross-lane traffic in the hot loop),
//         one 16-lane (m,s) butterfly merge at the end.
// Pass 2: att[q][m] = |p12 - p21| in bf16.
// LDS: a1s/a2s [c][q] (resident), b2sT/b1sT [m][c] so B-frags are 2x ds_load_b128.
// ---------------------------------------------------------------------------
__global__ __launch_bounds__(256) void attn_kernel(
    const bf16* __restrict__ ab1, const bf16* __restrict__ ab2,
    bf16* __restrict__ att, int N) {
  constexpr int BQ = 128, BMT = 128, CH = 64;
  constexpr int QP = BQ + 8;   // a-tile stride [c][q]
  constexpr int MPT = CH + 8;  // b-tile stride [m][c] = 72
  extern __shared__ bf16 sm[];
  bf16* a1s  = sm;                   // CH*QP
  bf16* a2s  = a1s + CH * QP;
  bf16* b2sT = a2s + CH * QP;        // BMT*MPT
  bf16* b1sT = b2sT + BMT * MPT;

  const int t = threadIdx.x, lane = t & 31, wid = t >> 5;
  const int half = lane >> 4, nc = lane & 15;
  const int q0 = blockIdx.x * BQ;

  {  // resident query blocks: a1s/a2s [c][q]
    const int c = t >> 2, qp = (t & 3) * 32;
    const uint4* s1 = (const uint4*)(ab1 + (size_t)c * N + q0 + qp);
    const uint4* s2 = (const uint4*)(ab2 + (size_t)c * N + q0 + qp);
    uint4* d1 = (uint4*)(a1s + c * QP + qp);
    uint4* d2 = (uint4*)(a2s + c * QP + qp);
#pragma unroll
    for (int i = 0; i < 4; ++i) { d1[i] = s1[i]; d2[i] = s2[i]; }
  }
  __syncthreads();

  const int mbase = wid * 16;
  v16bf fa1[2], fa2[2];
#pragma unroll
  for (int kk = 0; kk < 2; ++kk) {
    fa1[kk] = frag_a_colmajor(a1s, QP, mbase, kk * 32, lane);
    fa2[kk] = frag_a_colmajor(a2s, QP, mbase, kk * 32, lane);
  }

  // lane-local online stats for row (r + 8*half) over this lane's columns
  float rm12[8], rs12[8], rm21[8], rs21[8];
#pragma unroll
  for (int r = 0; r < 8; ++r) { rm12[r] = rm21[r] = -1e30f; rs12[r] = rs21[r] = 0.0f; }

  // b-tile loader: 4c x 8m micro-tile per thread: cg = t>>4 (0..15), mg = t&15
  const int cg = t >> 4, mg = t & 15;
  const int nmt = N / BMT;

  // ---------------- pass 1: logits + lane-local online softmax ----------------
  for (int mt = 0; mt < nmt; ++mt) {
    const int m0 = mt * BMT;
    {
      Q8 r2[4], r1[4];
#pragma unroll
      for (int i = 0; i < 4; ++i) {
        r2[i].u = *(const uint4*)(ab2 + (size_t)(CH + cg * 4 + i) * N + m0 + mg * 8);
        r1[i].u = *(const uint4*)(ab1 + (size_t)(CH + cg * 4 + i) * N + m0 + mg * 8);
      }
#pragma unroll
      for (int j = 0; j < 8; ++j) {
        Q4 c2, c1;
#pragma unroll
        for (int i = 0; i < 4; ++i) { c2.h[i] = r2[i].h[j]; c1.h[i] = r1[i].h[j]; }
        *(unsigned long long*)(&b2sT[(mg * 8 + j) * MPT + cg * 4]) = c2.u;
        *(unsigned long long*)(&b1sT[(mg * 8 + j) * MPT + cg * 4]) = c1.u;
      }
    }
    __syncthreads();
#pragma unroll
    for (int nt = 0; nt < 8; ++nt) {
      v8f s12 = (v8f)(0.0f), s21 = (v8f)(0.0f);
#pragma unroll
      for (int kk = 0; kk < 2; ++kk) {
        v16bf fb2 = frag_b_nk(b2sT, MPT, kk * 32, nt * 16, lane);
        v16bf fb1 = frag_b_nk(b1sT, MPT, kk * 32, nt * 16, lane);
        s12 = WMMA_BF16(fa1[kk], fb2, s12);
        s21 = WMMA_BF16(fa2[kk], fb1, s21);
      }
#pragma unroll
      for (int r = 0; r < 8; ++r) {
        const float x = s12[r];
        const float nm = fmaxf(rm12[r], x);
        rs12[r] = rs12[r] * __expf(rm12[r] - nm) + __expf(x - nm);
        rm12[r] = nm;
        const float y = s21[r];
        const float nm2 = fmaxf(rm21[r], y);
        rs21[r] = rs21[r] * __expf(rm21[r] - nm2) + __expf(y - nm2);
        rm21[r] = nm2;
      }
    }
    __syncthreads();
  }

  // ---------------- merge lane-local (m,s) pairs across the 16-lane group ----------------
#pragma unroll
  for (int r = 0; r < 8; ++r) {
#pragma unroll
    for (int mask = 1; mask <= 8; mask <<= 1) {
      float om = __shfl_xor(rm12[r], mask, 32);
      float os = __shfl_xor(rs12[r], mask, 32);
      float nm = fmaxf(rm12[r], om);
      rs12[r] = rs12[r] * __expf(rm12[r] - nm) + os * __expf(om - nm);
      rm12[r] = nm;
      float om2 = __shfl_xor(rm21[r], mask, 32);
      float os2 = __shfl_xor(rs21[r], mask, 32);
      float nm2 = fmaxf(rm21[r], om2);
      rs21[r] = rs21[r] * __expf(rm21[r] - nm2) + os2 * __expf(om2 - nm2);
      rm21[r] = nm2;
    }
  }
  float inv12[8], inv21[8];
#pragma unroll
  for (int r = 0; r < 8; ++r) { inv12[r] = 1.0f / rs12[r]; inv21[r] = 1.0f / rs21[r]; }

  // ---------------- pass 2: att = |p12 - p21| -> bf16 ----------------
  for (int mt = 0; mt < nmt; ++mt) {
    const int m0 = mt * BMT;
    {
      Q8 r2[4], r1[4];
#pragma unroll
      for (int i = 0; i < 4; ++i) {
        r2[i].u = *(const uint4*)(ab2 + (size_t)(CH + cg * 4 + i) * N + m0 + mg * 8);
        r1[i].u = *(const uint4*)(ab1 + (size_t)(CH + cg * 4 + i) * N + m0 + mg * 8);
      }
#pragma unroll
      for (int j = 0; j < 8; ++j) {
        Q4 c2, c1;
#pragma unroll
        for (int i = 0; i < 4; ++i) { c2.h[i] = r2[i].h[j]; c1.h[i] = r1[i].h[j]; }
        *(unsigned long long*)(&b2sT[(mg * 8 + j) * MPT + cg * 4]) = c2.u;
        *(unsigned long long*)(&b1sT[(mg * 8 + j) * MPT + cg * 4]) = c1.u;
      }
    }
    __syncthreads();
#pragma unroll
    for (int nt = 0; nt < 8; ++nt) {
      v8f s12 = (v8f)(0.0f), s21 = (v8f)(0.0f);
#pragma unroll
      for (int kk = 0; kk < 2; ++kk) {
        v16bf fb2 = frag_b_nk(b2sT, MPT, kk * 32, nt * 16, lane);
        v16bf fb1 = frag_b_nk(b1sT, MPT, kk * 32, nt * 16, lane);
        s12 = WMMA_BF16(fa1[kk], fb2, s12);
        s21 = WMMA_BF16(fa2[kk], fb1, s21);
      }
#pragma unroll
      for (int r = 0; r < 8; ++r) {
        const float p12 = __expf(s12[r] - rm12[r]) * inv12[r];
        const float p21 = __expf(s21[r] - rm21[r]) * inv21[r];
        const int row = q0 + mbase + r + 8 * half;
        const int col = m0 + nt * 16 + nc;
        att[(size_t)row * N + col] = (bf16)fabsf(p12 - p21);
      }
    }
    __syncthreads();
  }
}

// ---------------------------------------------------------------------------
extern "C" void kernel_launch(void* const* d_in, const int* in_sizes, int n_in,
                              void* d_out, int out_size, void* d_ws, size_t ws_size,
                              hipStream_t stream) {
  (void)in_sizes; (void)n_in; (void)out_size; (void)ws_size;
  const float* x1  = (const float*)d_in[0];
  const float* x2  = (const float*)d_in[1];
  const float* Wb  = (const float*)d_in[2];
  const float* bb  = (const float*)d_in[3];
  const float* Wc  = (const float*)d_in[4];
  const float* bc  = (const float*)d_in[5];
  const float* Wdd = (const float*)d_in[6];
  const float* bd  = (const float*)d_in[7];
  const float* al  = (const float*)d_in[8];
  const float* be  = (const float*)d_in[9];
  float* out = (float*)d_out;

  const int Bb = 2, C = 512, N = 4096;
  const size_t CN = (size_t)C * N;  // 2,097,152

  char* ws = (char*)d_ws;
  size_t off = 0;
  auto alloc = [&](size_t bytes) {
    char* p = ws + off;
    off = (off + bytes + 255) & ~(size_t)255;
    return p;
  };
  bf16*  x1b     = (bf16*)alloc(Bb * CN * sizeof(bf16));
  bf16*  x2b     = (bf16*)alloc(Bb * CN * sizeof(bf16));
  bf16*  xc      = (bf16*)alloc(Bb * CN * sizeof(bf16));
  bf16*  Wab     = (bf16*)alloc(128 * 512 * sizeof(bf16));
  bf16*  Wd      = (bf16*)alloc(512 * 512 * sizeof(bf16));
  float* bias_ab = (float*)alloc(128 * sizeof(float));
  float* bias_d  = (float*)alloc(512 * sizeof(float));
  bf16*  ab1     = (bf16*)alloc((size_t)Bb * 128 * N * sizeof(bf16));
  bf16*  ab2     = (bf16*)alloc((size_t)Bb * 128 * N * sizeof(bf16));
  bf16*  fc      = (bf16*)alloc(Bb * CN * sizeof(bf16));
  bf16*  att     = (bf16*)alloc((size_t)N * N * sizeof(bf16));  // reused per batch

  // 1) convert inputs + fold alpha/beta into xc
  const int total4 = (int)(Bb * CN / 4);
  pack_kernel<<<(total4 + 255) / 256, 256, 0, stream>>>(x1, x2, al, be, x1b, x2b, xc, total4);
  prep_weights<<<(512 * 512 + 255) / 256, 256, 0, stream>>>(Wb, Wc, Wdd, bb, bc, bd, al, be,
                                                            Wab, Wd, bias_ab, bias_d);

  // 2) projection GEMMs (bf16 out): ab = [Wb;Wc] @ x + [bb;bc];  fc = Wd @ xc + (a+b)*bd
  for (int b = 0; b < Bb; ++b) {
    gemm128<false, bf16><<<dim3(N / 128, 1), 256, 0, stream>>>(
        Wab, 512, x1b + b * CN, N, ab1 + (size_t)b * 128 * N, N, 512, bias_ab);
    gemm128<false, bf16><<<dim3(N / 128, 1), 256, 0, stream>>>(
        Wab, 512, x2b + b * CN, N, ab2 + (size_t)b * 128 * N, N, 512, bias_ab);
    gemm128<false, bf16><<<dim3(N / 128, C / 128), 256, 0, stream>>>(
        Wd, 512, xc + b * CN, N, fc + b * CN, N, 512, bias_d);
  }

  // 3) attention diff (32MB att buffer reused; stream order serializes batches)
  const size_t attn_lds =
      (size_t)(2 * 64 * (128 + 8) + 2 * 128 * (64 + 8)) * sizeof(bf16);  // 71,680 B
  for (int b = 0; b < Bb; ++b) {
    attn_kernel<<<N / 128, 256, attn_lds, stream>>>(
        ab1 + (size_t)b * 128 * N, ab2 + (size_t)b * 128 * N, att, N);
    // 4) out[b] = fc[b] @ att^T  (fp32 epilogue straight to d_out)
    gemm128<true, float><<<dim3(N / 128, C / 128), 256, 0, stream>>>(
        fc + b * CN, N, att, N, out + b * CN, N, N, nullptr);
  }
}
